// MultiHeadAttention_12317966205504
// MI455X (gfx1250) — matrix operand order repaired and checked by reading, not compiled, not run on hardware
//
#include <hip/hip_runtime.h>

// ---------------------------------------------------------------------------
// T5-style multi-head attention for MI455X (gfx1250), bf16 WMMA pipeline.
//   B=2, S=2048, D=768, H=12, hd=64.
// Stage 1: f32 -> bf16 converts (activations) + transpose-convert (weights)
// Stage 2: WMMA GEMM, one wave = 16x64 strip (A-fragment reused 4x)
//          Q,K -> [B,H,S,64]; V -> [B,H,64,S] (transposed)
// Stage 3: flash attention, 4 waves/block share one (b,h); K/V tiles staged
//          into LDS by the Tensor Data Mover (tensor_load_to_lds +
//          s_wait_tensorcnt), softmax online, P restaged via LDS for WMMA.
// Stage 4: WMMA GEMM output projection -> d_out (f32).
// ---------------------------------------------------------------------------

#define DM   768
#define NH   12
#define HD   64
#define SEQ  2048
#define BATCH 2
#define MTOT (BATCH * SEQ)      // 4096 rows

typedef __bf16 bf16_t;
typedef bf16_t v16bf __attribute__((ext_vector_type(16)));
typedef float  v8f   __attribute__((ext_vector_type(8)));
typedef unsigned int v4u __attribute__((ext_vector_type(4)));
typedef int          v8i __attribute__((ext_vector_type(8)));
typedef int          v4i __attribute__((ext_vector_type(4)));

#if __has_builtin(__builtin_amdgcn_tensor_load_to_lds) && \
    __has_builtin(__builtin_amdgcn_s_wait_tensorcnt)
#define USE_TDM 1
#else
#define USE_TDM 0
#endif

union Frag16 {
    v16bf v;
    uint4 u[2];
};

__device__ __forceinline__ unsigned short f2bf(float x) {
    unsigned int u = __float_as_uint(x);
    u += 0x7FFFu + ((u >> 16) & 1u);      // round-to-nearest-even
    return (unsigned short)(u >> 16);
}

// ---------------------------------------------------------------------------
// Stage 1 kernels
// ---------------------------------------------------------------------------
__global__ void cvt_f32_bf16(const float* __restrict__ in,
                             unsigned short* __restrict__ out, int n) {
    int i = blockIdx.x * blockDim.x + threadIdx.x;
    if (i < n) out[i] = f2bf(in[i]);
}

// Wt[n][k] = W[k][n], converted to bf16 (so WMMA B-fragment loads are rows).
__global__ void transpose_w(const float* __restrict__ W,
                            unsigned short* __restrict__ Wt) {
    int i = blockIdx.x * blockDim.x + threadIdx.x;   // over 768*768
    if (i >= DM * DM) return;
    int n = i / DM, k = i % DM;
    Wt[i] = f2bf(W[(size_t)k * DM + n]);
}

// ---------------------------------------------------------------------------
// Stage 2/4: bf16 WMMA GEMM. One wave computes a 16x64 strip of
// C[4096,768] = A[4096,768] * W[768,768]; A fragment reused across 4 N-tiles.
// mode 0: write bf16 [B,H,S,64]   (Q, K)     (strip ng == head)
// mode 1: write bf16 [B,H,64,S]   (V transposed)
// mode 2: write f32  [M,768]      (final output)
// ---------------------------------------------------------------------------
__global__ void __launch_bounds__(256)
gemm_bf16(const unsigned short* __restrict__ A,
          const unsigned short* __restrict__ Wt,
          void* __restrict__ outp, int mode) {
    const int lane = threadIdx.x & 31;
    const int wid  = threadIdx.x >> 5;
    const int l16  = lane & 15;
    const int half = lane >> 4;

    int strip = blockIdx.x * 8 + wid;         // (4096/16)*(768/64) = 3072
    int mt = strip / (DM / 64);               // 0..255
    int ng = strip % (DM / 64);               // 0..11 (== head for mode 0/1)

    const unsigned short* arow = A + (size_t)(mt * 16 + l16) * DM;
    const int ao = half ? 16 : 0;   // A frag: K 0..15 / 16..31
    const int bo = half ? 16 : 0;   // B frag: K 0..15 / 16..31

    v8f acc[4] = {};
    for (int k = 0; k < DM; k += 32) {
        Frag16 a;
        a.u[0] = *(const uint4*)(arow + k + ao);
        a.u[1] = *(const uint4*)(arow + k + ao + 8);
        #pragma unroll
        for (int j = 0; j < 4; ++j) {
            const unsigned short* brow =
                Wt + (size_t)(ng * 64 + j * 16 + l16) * DM + k + bo;
            Frag16 b;
            b.u[0] = *(const uint4*)(brow);
            b.u[1] = *(const uint4*)(brow + 8);
            acc[j] = __builtin_amdgcn_wmma_f32_16x16x32_bf16(
                         false, a.v, false, b.v, (short)0, acc[j], false, false);
        }
    }

    if (mode == 2) {
        float* out = (float*)outp;
        #pragma unroll
        for (int j = 0; j < 4; ++j)
            #pragma unroll
            for (int r = 0; r < 8; ++r) {
                int m = mt * 16 + r + 8 * half;
                out[(size_t)m * DM + ng * 64 + j * 16 + l16] = acc[j][r];
            }
    } else {
        unsigned short* out = (unsigned short*)outp;
        #pragma unroll
        for (int j = 0; j < 4; ++j)
            #pragma unroll
            for (int r = 0; r < 8; ++r) {
                int m = mt * 16 + r + 8 * half;
                int b = m >> 11, s = m & (SEQ - 1);
                int d = j * 16 + l16;
                size_t off = (mode == 0)
                    ? ((((size_t)b * NH + ng) * SEQ + s) * HD + d)
                    : ((((size_t)b * NH + ng) * HD + d) * SEQ + s);
                out[off] = f2bf(acc[j][r]);
            }
    }
}

// ---------------------------------------------------------------------------
// Tensor Data Mover: 2D bf16 tile (rows x row_len, row stride in elements)
// from global memory into LDS.  D# built per CDNA5 ISA 8.3/8.4.
// This toolchain's builtin takes 6 args (g0, g1, g2, g3, g1-ext, cpol).
// ---------------------------------------------------------------------------
#if USE_TDM
__device__ __forceinline__ void tdm_load_2d(unsigned lds_addr, const void* gptr,
                                            unsigned row_len, unsigned rows,
                                            unsigned stride) {
    unsigned long long ga = (unsigned long long)(uintptr_t)gptr;
    v4u g0 = { 1u,                                   // count=1, no gather
               lds_addr,                             // lds_addr [63:32]
               (unsigned)ga,                         // global_addr low
               (unsigned)((ga >> 32) & 0x01FFFFFFu)  // global_addr [56:32]
                   | 0x80000000u };                  // type=2 ("image")
    v8i g1 = { (int)0x00010000u,          // data_size=1 (2 bytes)
               (int)(row_len << 16),      // tensor_dim0[15:0]   (bits 63:48)
               (int)(rows << 16),         // tensor_dim1[15:0]   (bits 95:80)
               (int)(row_len << 16),      // tile_dim0           (bits 127:112)
               (int)rows,                 // tile_dim1           (bits 143:128)
               (int)stride,               // tensor_dim0_stride low 32
               0, 0 };
    v4i z4 = {};
    v8i z8 = {};
    __builtin_amdgcn_tensor_load_to_lds(g0, g1, z4, z4, z8, 0);
}
#endif

// ---------------------------------------------------------------------------
// Stage 3: flash attention.
// ---------------------------------------------------------------------------
__device__ __forceinline__ float rmax16(float x) {
    x = fmaxf(x, __shfl_xor(x, 1));
    x = fmaxf(x, __shfl_xor(x, 2));
    x = fmaxf(x, __shfl_xor(x, 4));
    x = fmaxf(x, __shfl_xor(x, 8));
    return x;   // masks <16 stay within each 16-lane half of the wave
}
__device__ __forceinline__ float rsum16(float x) {
    x += __shfl_xor(x, 1);
    x += __shfl_xor(x, 2);
    x += __shfl_xor(x, 4);
    x += __shfl_xor(x, 8);
    return x;
}

// T5 bidirectional relative-position bucket bias, computed per element.
__device__ __forceinline__ float rel_bias(int qg, int key, int h,
                                          const float* __restrict__ rpb) {
    int n = qg - key;
    int base = 0;
    if (n < 0) { base = 16; n = -n; }
    float v;
    if (n < 8) {
        v = (float)n;
    } else {
        float large = 8.0f +
            __logf((float)n * 0.125f + 1e-6f) * (8.0f / 2.772588722239781f);
        v = fminf(large, 15.0f);
    }
    int bucket = base + (int)v;
    return rpb[bucket * NH + h];
}

__global__ void __launch_bounds__(128)
attn_flash(const unsigned short* __restrict__ Qh,   // [B,H,S,64] bf16
           const unsigned short* __restrict__ Kh,   // [B,H,S,64] bf16
           const unsigned short* __restrict__ Vt,   // [B,H,64,S] bf16
           const float* __restrict__ mask,          // [B,1,S,S] f32
           const float* __restrict__ rpb,           // [32,12] f32
           unsigned short* __restrict__ Obf) {      // [B,S,768] bf16
    // Shared K tile [32 keys][64 dims], V tile [64 dims][32 keys],
    // and per-wave P staging (16x32 each).
    __shared__ __align__(16) unsigned short lds_k[32 * 64];
    __shared__ __align__(16) unsigned short lds_v[64 * 32];
    __shared__ __align__(16) unsigned short pstage[4 * 16 * 32];

    const int lane = threadIdx.x & 31;
    const int wid  = threadIdx.x >> 5;
    const int l16  = lane & 15;
    const int half = lane >> 4;

    // 4 waves per block share one (b,h) and take 4 consecutive q tiles.
    int bh = blockIdx.x / 32;                 // 0..23
    int qt = (blockIdx.x % 32) * 4 + wid;     // 0..127
    int h  = bh % NH, b = bh / NH;

    const size_t bhoff = (size_t)bh * SEQ * HD;
    const int ao = half ? 16 : 0;
    const int ko = half ? 16 : 0;

    // Resident Q fragments (16 rows x 64 dims -> two 16x32 A frags)
    Frag16 q0, q1;
    {
        const unsigned short* qrow = Qh + bhoff + (size_t)(qt * 16 + l16) * HD;
        q0.u[0] = *(const uint4*)(qrow + ao);
        q0.u[1] = *(const uint4*)(qrow + ao + 8);
        q1.u[0] = *(const uint4*)(qrow + 32 + ao);
        q1.u[1] = *(const uint4*)(qrow + 32 + ao + 8);
    }

    float mrow[8], lrow[8];
    v8f oacc[4] = {};
    #pragma unroll
    for (int r = 0; r < 8; ++r) { mrow[r] = -1e30f; lrow[r] = 0.0f; }

    unsigned short* myp = pstage + wid * 512;

    for (int kb = 0; kb < SEQ; kb += 32) {
        // ---- stage shared K/V tiles into LDS
#if USE_TDM
        if (wid == 0) {
            tdm_load_2d((unsigned)(uintptr_t)lds_k,
                        Kh + bhoff + (size_t)kb * HD, HD, 32, HD);
            tdm_load_2d((unsigned)(uintptr_t)lds_v,
                        Vt + bhoff + kb, 32, HD, SEQ);
            __builtin_amdgcn_s_wait_tensorcnt(0);
        }
#else
        {   // cooperative fallback copy
            const unsigned* ksrc = (const unsigned*)(Kh + bhoff + (size_t)kb * HD);
            for (int i = threadIdx.x; i < 512; i += 128)
                ((unsigned*)lds_k)[i] = ksrc[i];
            int i = threadIdx.x;                      // 128 uint4 = half of V tile
            int row = i >> 1, part = i & 1;
            ((uint4*)lds_v)[i] =
                *(const uint4*)(Vt + bhoff + (size_t)row * SEQ + kb + part * 8);
        }
#endif
        __syncthreads();

        // ---- scores: Q[16,64] @ K^T[64,32] -> two 16x16 f32 fragments
        v8f c0 = {}, c1 = {};
        {
            const unsigned short* kp0 = lds_k + (size_t)l16 * HD + ko;
            const unsigned short* kp1 = lds_k + (size_t)(16 + l16) * HD + ko;
            Frag16 ka, kbb;
            ka.u[0]  = *(const uint4*)(kp0);      ka.u[1]  = *(const uint4*)(kp0 + 8);
            kbb.u[0] = *(const uint4*)(kp0 + 32); kbb.u[1] = *(const uint4*)(kp0 + 40);
            c0 = __builtin_amdgcn_wmma_f32_16x16x32_bf16(false, q0.v, false, ka.v,  (short)0, c0, false, false);
            c0 = __builtin_amdgcn_wmma_f32_16x16x32_bf16(false, q1.v, false, kbb.v, (short)0, c0, false, false);
            ka.u[0]  = *(const uint4*)(kp1);      ka.u[1]  = *(const uint4*)(kp1 + 8);
            kbb.u[0] = *(const uint4*)(kp1 + 32); kbb.u[1] = *(const uint4*)(kp1 + 40);
            c1 = __builtin_amdgcn_wmma_f32_16x16x32_bf16(false, q0.v, false, ka.v,  (short)0, c1, false, false);
            c1 = __builtin_amdgcn_wmma_f32_16x16x32_bf16(false, q1.v, false, kbb.v, (short)0, c1, false, false);
        }

        // ---- scale + rel-pos bias + mask, online softmax update
        float p0[8], p1[8];
        #pragma unroll
        for (int r = 0; r < 8; ++r) {
            int qg = qt * 16 + r + 8 * half;
            size_t moff = ((size_t)b * SEQ + qg) * SEQ + kb;
            if (r == 0 && kb + 32 < SEQ)
                __builtin_prefetch(mask + moff + 32, 0, 0);
            float mk0 = mask[moff + l16];
            float mk1 = mask[moff + 16 + l16];
            float s0 = c0[r] * 0.125f + rel_bias(qg, kb + l16, h, rpb)
                     + (1.0f - mk0) * -1e9f;
            float s1 = c1[r] * 0.125f + rel_bias(qg, kb + 16 + l16, h, rpb)
                     + (1.0f - mk1) * -1e9f;

            float tm = rmax16(fmaxf(s0, s1));
            float mn = fmaxf(mrow[r], tm);
            float sc = __expf(mrow[r] - mn);
            p0[r] = __expf(s0 - mn);
            p1[r] = __expf(s1 - mn);
            float ts = rsum16(p0[r] + p1[r]);
            lrow[r] = lrow[r] * sc + ts;
            mrow[r] = mn;
            oacc[0][r] *= sc; oacc[1][r] *= sc;
            oacc[2][r] *= sc; oacc[3][r] *= sc;
        }

        // ---- stage P (16x32) to LDS, reload as 16x32 bf16 A-fragment
        #pragma unroll
        for (int r = 0; r < 8; ++r) {
            int row = r + 8 * half;
            myp[row * 32 + l16]      = f2bf(p0[r]);
            myp[row * 32 + 16 + l16] = f2bf(p1[r]);
        }
        Frag16 pf;
        pf.u[0] = *(const uint4*)(myp + l16 * 32 + ao);
        pf.u[1] = *(const uint4*)(myp + l16 * 32 + ao + 8);

        // ---- O += P[16,32] @ V[32,64]  (four 16x16 output fragments)
        #pragma unroll
        for (int j = 0; j < 4; ++j) {
            const unsigned short* vp = lds_v + (size_t)(j * 16 + l16) * 32 + ko;
            Frag16 vf;
            vf.u[0] = *(const uint4*)(vp);
            vf.u[1] = *(const uint4*)(vp + 8);
            oacc[j] = __builtin_amdgcn_wmma_f32_16x16x32_bf16(
                          false, pf.v, false, vf.v, (short)0, oacc[j], false, false);
        }
        __syncthreads();   // before next TDM overwrite of lds_k/lds_v
    }

    // ---- epilogue: normalize and store merged-head bf16 activation
    #pragma unroll
    for (int j = 0; j < 4; ++j) {
        #pragma unroll
        for (int r = 0; r < 8; ++r) {
            int qg = qt * 16 + r + 8 * half;
            float v = oacc[j][r] / lrow[r];
            Obf[((size_t)b * SEQ + qg) * DM + h * HD + j * 16 + l16] = f2bf(v);
        }
    }
}

// ---------------------------------------------------------------------------
// Host-side launcher
// ---------------------------------------------------------------------------
extern "C" void kernel_launch(void* const* d_in, const int* in_sizes, int n_in,
                              void* d_out, int out_size, void* d_ws, size_t ws_size,
                              hipStream_t stream) {
    (void)in_sizes; (void)n_in; (void)out_size; (void)ws_size;

    const float* q    = (const float*)d_in[0];
    const float* k    = (const float*)d_in[1];
    const float* v    = (const float*)d_in[2];
    const float* mask = (const float*)d_in[3];
    const float* Wq   = (const float*)d_in[4];
    const float* Wk   = (const float*)d_in[5];
    const float* Wv   = (const float*)d_in[6];
    const float* Wo   = (const float*)d_in[7];
    const float* rpb  = (const float*)d_in[8];
    float* out = (float*)d_out;

    // Workspace layout (bf16 = unsigned short), ~49 MB total.
    const size_t ACT  = (size_t)MTOT * DM;   // 3,145,728
    const size_t WSZ  = (size_t)DM * DM;     //   589,824
    unsigned short* ws  = (unsigned short*)d_ws;
    unsigned short* qbf = ws;
    unsigned short* kbf = qbf + ACT;
    unsigned short* vbf = kbf + ACT;
    unsigned short* wt  = vbf + ACT;         // 4 transposed weights
    unsigned short* Qh  = wt + 4 * WSZ;
    unsigned short* Kh  = Qh + ACT;
    unsigned short* Vt  = Kh + ACT;
    unsigned short* Obf = Vt + ACT;

    // Stage 1: conversions
    {
        int n = (int)ACT;
        int g = (n + 255) / 256;
        cvt_f32_bf16<<<g, 256, 0, stream>>>(q, qbf, n);
        cvt_f32_bf16<<<g, 256, 0, stream>>>(k, kbf, n);
        cvt_f32_bf16<<<g, 256, 0, stream>>>(v, vbf, n);
        int gw = (int)((WSZ + 255) / 256);
        transpose_w<<<gw, 256, 0, stream>>>(Wq, wt + 0 * WSZ);
        transpose_w<<<gw, 256, 0, stream>>>(Wk, wt + 1 * WSZ);
        transpose_w<<<gw, 256, 0, stream>>>(Wv, wt + 2 * WSZ);
        transpose_w<<<gw, 256, 0, stream>>>(Wo, wt + 3 * WSZ);
    }

    // Stage 2: QKV projections. strips = 256*12 = 3072; 8 waves/block
    const int gemm_blocks = (MTOT / 16) * (DM / 64) / 8;   // 384
    gemm_bf16<<<gemm_blocks, 256, 0, stream>>>(qbf, wt + 0 * WSZ, Qh, 0);
    gemm_bf16<<<gemm_blocks, 256, 0, stream>>>(kbf, wt + 1 * WSZ, Kh, 0);
    gemm_bf16<<<gemm_blocks, 256, 0, stream>>>(vbf, wt + 2 * WSZ, Vt, 1);

    // Stage 3: flash attention. 24 (b,h) * 32 blocks, 4 q-tiles per block.
    attn_flash<<<BATCH * NH * 32, 128, 0, stream>>>(Qh, Kh, Vt, mask, rpb, Obf);

    // Stage 4: output projection -> f32 d_out
    gemm_bf16<<<gemm_blocks, 256, 0, stream>>>(Obf, wt + 3 * WSZ, out, 2);
}
